// SimpleMLA_14877766713422
// MI455X (gfx1250) — matrix operand order repaired and checked by reading, not compiled, not run on hardware
//
#include <hip/hip_runtime.h>
#include <hip/hip_bf16.h>

typedef __bf16 bf16_t;
typedef __attribute__((ext_vector_type(16))) __bf16 v16bf;
typedef __attribute__((ext_vector_type(8)))  float  v8f;
typedef uint32_t u32x4 __attribute__((ext_vector_type(4)));
typedef uint32_t u32x8 __attribute__((ext_vector_type(8)));

// ---------------------------------------------------------------------------
// WMMA helpers (CDNA5 16x16x32 bf16, fp32 accumulate)
// ---------------------------------------------------------------------------
__device__ inline v8f wmma_bf16(v16bf a, v16bf b, v8f c) {
  return __builtin_amdgcn_wmma_f32_16x16x32_bf16(
      /*neg_a=*/false, a, /*neg_b=*/false, b,
      /*c_mod=*/(short)0, c, /*reuse_a=*/false, /*reuse_b=*/false);
}

// A fragment: 16x32 (MxK), row-major source. ISA layout:
// lane L: M = L%16, half = L/16; elem j(0..7) -> K = half*8 + j,
// elem j(8..15) -> K = 16 + half*8 + (j-8).
template <typename T>
__device__ inline v16bf load_a_frag(const T* __restrict__ base, int ld) {
  const int lane = threadIdx.x & 31;
  const int m = lane & 15, half = lane >> 4;
  const T* r0 = base + (long)m * ld + half * 8;
  const T* r1 = r0 + 16;
  v16bf a;
#pragma unroll
  for (int j = 0; j < 8; ++j) a[j] = (__bf16)(float)r0[j];
#pragma unroll
  for (int j = 0; j < 8; ++j) a[8 + j] = (__bf16)(float)r1[j];
  return a;
}

// B fragment: 32x16 (KxN), loaded from BT (N-major, K contiguous per row).
// lane L: N = L%16, kb = (L/16)*8; elem j(0..7) -> K = kb + j,
// elem j(8..15) -> K = 16 + kb + (j-8).
template <typename T>
__device__ inline v16bf load_bt_frag(const T* __restrict__ bt, int ld) {
  const int lane = threadIdx.x & 31;
  const int n = lane & 15, kb = (lane >> 4) * 8;
  const T* r = bt + (long)n * ld + kb;
  v16bf b;
#pragma unroll
  for (int j = 0; j < 8; ++j) b[j] = (__bf16)(float)r[j];
#pragma unroll
  for (int j = 0; j < 8; ++j) b[8 + j] = (__bf16)(float)r[16 + j];
  return b;
}

// C/D fragment layout: lane L: N = L%16, rows M = 8*(L/16) + r for VGPR r.
__device__ inline void store_c_f32(float* c, long ld, v8f acc) {
  const int lane = threadIdx.x & 31;
  const int n = lane & 15, mb = (lane >> 4) * 8;
#pragma unroll
  for (int r = 0; r < 8; ++r) c[(long)(mb + r) * ld + n] = acc[r];
}
__device__ inline void store_c_bf16(bf16_t* c, long ld, v8f acc) {
  const int lane = threadIdx.x & 31;
  const int n = lane & 15, mb = (lane >> 4) * 8;
#pragma unroll
  for (int r = 0; r < 8; ++r) c[(long)(mb + r) * ld + n] = (__bf16)acc[r];
}
// Transposed store: out[(n0+n)*ld + (m0+m)]
__device__ inline void store_cT_bf16(bf16_t* c, long ld, int m0, int n0, v8f acc) {
  const int lane = threadIdx.x & 31;
  const int n = lane & 15, mb = (lane >> 4) * 8;
#pragma unroll
  for (int r = 0; r < 8; ++r) c[(long)(n0 + n) * ld + m0 + mb + r] = (__bf16)acc[r];
}

// ---------------------------------------------------------------------------
// Tensor Data Mover: async 2D tile load global -> LDS.
// Builds D# group0 (count/lds_addr/global_addr/type=2) + group1
// (mask=0, data_size=2B, tensor dims == tile dims, dim0 stride) in SGPR
// tuples and issues TENSOR_LOAD_TO_LDS (2D: VADDR2/3 = NULL form).
// All inputs must be wave-uniform. Tracked by TENSORcnt.
// ---------------------------------------------------------------------------
__device__ inline void tdm_load_2d(bf16_t* lds_dst, const bf16_t* gsrc,
                                   long ld_elems, int rows, int cols) {
  const uint32_t lofs = (uint32_t)(size_t)lds_dst;  // addr[31:0] = LDS offset
  const uint64_t ga = (uint64_t)(size_t)gsrc;
  u32x4 g0;
  g0[0] = 1u;                                                // count=1
  g0[1] = lofs;                                              // lds_addr
  g0[2] = (uint32_t)ga;                                      // global_addr lo
  g0[3] = (uint32_t)((ga >> 32) & 0x01FFFFFFu) | (2u << 30); // ga hi | type=2
  const uint32_t td0 = (uint32_t)cols;   // tensor_dim0 (elements)
  const uint32_t td1 = (uint32_t)rows;   // tensor_dim1
  const uint64_t st0 = (uint64_t)ld_elems;
  u32x8 g1;
  g1[0] = (1u << 16);                                   // data_size=1 (2 bytes)
  g1[1] = (td0 & 0xFFFFu) << 16;                        // tensor_dim0[15:0]
  g1[2] = (td0 >> 16) | ((td1 & 0xFFFFu) << 16);        // td0 hi | td1 lo
  g1[3] = (td1 >> 16) | (((uint32_t)cols) << 16);       // td1 hi | tile_dim0
  g1[4] = (uint32_t)rows & 0xFFFFu;                     // tile_dim1 (dim2=0)
  g1[5] = (uint32_t)(st0 & 0xFFFFFFFFu);                // dim0_stride[31:0]
  g1[6] = (uint32_t)((st0 >> 32) & 0xFFFFu);            // dim0_stride[47:32]
  g1[7] = 0u;
  asm volatile("tensor_load_to_lds %0, %1" :: "s"(g0), "s"(g1) : "memory");
}

// ---------------------------------------------------------------------------
// Elementwise conversion / transpose kernels
// ---------------------------------------------------------------------------
__global__ void mla_cvt_bf16_kernel(const float* __restrict__ src,
                                    bf16_t* __restrict__ dst, long n) {
  long i = (long)blockIdx.x * blockDim.x + threadIdx.x;
  const long stride = (long)gridDim.x * blockDim.x;
  for (; i < n; i += stride) dst[i] = (__bf16)src[i];
}

// dst[c * R + r] = src[r * C + c]   (R x C fp32 -> C x R bf16)
__global__ void mla_transpose_cvt_kernel(const float* __restrict__ src,
                                         bf16_t* __restrict__ dst,
                                         int R, int C) {
  long i = (long)blockIdx.x * blockDim.x + threadIdx.x;
  const long stride = (long)gridDim.x * blockDim.x;
  const long n = (long)R * C;
  for (; i < n; i += stride) {
    const int r = (int)(i / C), c = (int)(i % C);
    dst[(long)c * R + r] = (__bf16)src[i];
  }
}

// ---------------------------------------------------------------------------
// Generic batched GEMM (direct L2 path): C[m,n] = sum_k A[m,k] * BT[n,k]
// Each wave computes a 32x32 tile (2x2 fragments, 4 WMMAs per K step).
// mode: 0 = f32 store, 1 = bf16 store, 2 = transposed bf16 store.
// ---------------------------------------------------------------------------
__global__ __launch_bounds__(256) void mla_gemm_bt_kernel(
    const bf16_t* __restrict__ A, long lda, long sAb, long sAh,
    const bf16_t* __restrict__ BT, long ldb, long sBb, long sBh,
    void* __restrict__ Cp, long ldc, long sCb, long sCh,
    int M, int N, int K, int HB, int mode) {
  const int wave = threadIdx.x >> 5;
  const int z = blockIdx.y;
  const int b = z / HB, h = z % HB;
  A += b * sAb + h * sAh;
  BT += b * sBb + h * sBh;
  const long coff = b * sCb + h * sCh;

  const int mt = M >> 5;
  const int tiles = mt * (N >> 5);
  const int t = blockIdx.x * 8 + wave;
  if (t >= tiles) return;
  const int m0 = (t % mt) * 32, n0 = (t / mt) * 32;

  v8f a00{}, a01{}, a10{}, a11{};
  for (int k0 = 0; k0 < K; k0 += 32) {
    const v16bf fa0 = load_a_frag(A + (long)m0 * lda + k0, (int)lda);
    const v16bf fa1 = load_a_frag(A + (long)(m0 + 16) * lda + k0, (int)lda);
    const v16bf fb0 = load_bt_frag(BT + (long)n0 * ldb + k0, (int)ldb);
    const v16bf fb1 = load_bt_frag(BT + (long)(n0 + 16) * ldb + k0, (int)ldb);
    a00 = wmma_bf16(fa0, fb0, a00);
    a01 = wmma_bf16(fa0, fb1, a01);
    a10 = wmma_bf16(fa1, fb0, a10);
    a11 = wmma_bf16(fa1, fb1, a11);
  }
  if (mode == 0) {
    float* C = (float*)Cp + coff;
    store_c_f32(C + (long)m0 * ldc + n0, ldc, a00);
    store_c_f32(C + (long)m0 * ldc + n0 + 16, ldc, a01);
    store_c_f32(C + (long)(m0 + 16) * ldc + n0, ldc, a10);
    store_c_f32(C + (long)(m0 + 16) * ldc + n0 + 16, ldc, a11);
  } else if (mode == 1) {
    bf16_t* C = (bf16_t*)Cp + coff;
    store_c_bf16(C + (long)m0 * ldc + n0, ldc, a00);
    store_c_bf16(C + (long)m0 * ldc + n0 + 16, ldc, a01);
    store_c_bf16(C + (long)(m0 + 16) * ldc + n0, ldc, a10);
    store_c_bf16(C + (long)(m0 + 16) * ldc + n0 + 16, ldc, a11);
  } else {
    bf16_t* C = (bf16_t*)Cp + coff;
    store_cT_bf16(C, ldc, m0, n0, a00);
    store_cT_bf16(C, ldc, m0, n0 + 16, a01);
    store_cT_bf16(C, ldc, m0 + 16, n0, a10);
    store_cT_bf16(C, ldc, m0 + 16, n0 + 16, a11);
  }
}

// ---------------------------------------------------------------------------
// TDM-staged GEMM for the large projections.
// Block = 8 waves = 128x64 macro-tile; K in 64-element slabs, double-buffered
// in LDS via TENSOR_LOAD_TO_LDS (each wave DMAs 16 A-rows + 8 BT-rows, waits
// its own TENSORcnt, then workgroup barrier publishes the buffer).
// mode: 0 = f32 store, 2 = transposed bf16 store.
// ---------------------------------------------------------------------------
__global__ __launch_bounds__(256) void mla_gemm_tdm_kernel(
    const bf16_t* __restrict__ A, long lda, long sAb,
    const bf16_t* __restrict__ BT, long ldb, long sBb,
    void* __restrict__ Cp, long ldc, long sCb,
    int M, int N, int K, int mode) {
  __shared__ bf16_t Abuf[2][128 * 64];  // 32 KB
  __shared__ bf16_t Bbuf[2][64 * 64];   // 16 KB
  const int wid = __builtin_amdgcn_readfirstlane(threadIdx.x >> 5);
  const int z = blockIdx.y;
  A += z * sAb;
  BT += z * sBb;
  const long coff = z * sCb;

  const int mtiles = M / 128;
  const int m0 = (blockIdx.x % mtiles) * 128;
  const int n0 = (blockIdx.x / mtiles) * 64;
  const int wm = (wid & 3) * 32, wn = (wid >> 2) * 32;

  auto issue = [&](int buf, int k0) {
    tdm_load_2d(&Abuf[buf][wid * 16 * 64],
                A + (long)(m0 + wid * 16) * lda + k0, lda, 16, 64);
    tdm_load_2d(&Bbuf[buf][wid * 8 * 64],
                BT + (long)(n0 + wid * 8) * ldb + k0, ldb, 8, 64);
  };

  issue(0, 0);
  __builtin_amdgcn_s_wait_tensorcnt(0);
  __syncthreads();

  v8f a00{}, a01{}, a10{}, a11{};
  int buf = 0;
  for (int k0 = 0; k0 < K; k0 += 64) {
    if (k0 + 64 < K) issue(buf ^ 1, k0 + 64);
#pragma unroll
    for (int kk = 0; kk < 64; kk += 32) {
      const v16bf fa0 = load_a_frag(&Abuf[buf][wm * 64 + kk], 64);
      const v16bf fa1 = load_a_frag(&Abuf[buf][(wm + 16) * 64 + kk], 64);
      const v16bf fb0 = load_bt_frag(&Bbuf[buf][wn * 64 + kk], 64);
      const v16bf fb1 = load_bt_frag(&Bbuf[buf][(wn + 16) * 64 + kk], 64);
      a00 = wmma_bf16(fa0, fb0, a00);
      a01 = wmma_bf16(fa0, fb1, a01);
      a10 = wmma_bf16(fa1, fb0, a10);
      a11 = wmma_bf16(fa1, fb1, a11);
    }
    __builtin_amdgcn_s_wait_tensorcnt(0);
    __syncthreads();
    buf ^= 1;
  }

  if (mode == 0) {
    float* C = (float*)Cp + coff;
    store_c_f32(C + (long)(m0 + wm) * ldc + n0 + wn, ldc, a00);
    store_c_f32(C + (long)(m0 + wm) * ldc + n0 + wn + 16, ldc, a01);
    store_c_f32(C + (long)(m0 + wm + 16) * ldc + n0 + wn, ldc, a10);
    store_c_f32(C + (long)(m0 + wm + 16) * ldc + n0 + wn + 16, ldc, a11);
  } else {
    bf16_t* C = (bf16_t*)Cp + coff;
    store_cT_bf16(C, ldc, m0 + wm, n0 + wn, a00);
    store_cT_bf16(C, ldc, m0 + wm, n0 + wn + 16, a01);
    store_cT_bf16(C, ldc, m0 + wm + 16, n0 + wn, a10);
    store_cT_bf16(C, ldc, m0 + wm + 16, n0 + wn + 16, a11);
  }
}

// ---------------------------------------------------------------------------
// c = LayerNorm(x @ W_dkv^T): one block = 16 rows x 512 cols.
// ---------------------------------------------------------------------------
__global__ __launch_bounds__(256) void mla_c_ln_kernel(
    const bf16_t* __restrict__ xbf, const bf16_t* __restrict__ wdkv,
    const float* __restrict__ ln_w, const float* __restrict__ ln_b,
    float* __restrict__ c_out, bf16_t* __restrict__ c_bf) {
  __shared__ float tile[16 * 512];
  const int wave = threadIdx.x >> 5;
  const int lane = threadIdx.x & 31;
  const int m0 = blockIdx.x * 16;
  const bf16_t* Arow = xbf + (long)m0 * 2048;
  const int n0 = wave * 64;

  v8f acc[4] = {};
  for (int k0 = 0; k0 < 2048; k0 += 32) {
    const v16bf fa = load_a_frag(Arow + k0, 2048);
#pragma unroll
    for (int j = 0; j < 4; ++j) {
      const v16bf fb = load_bt_frag(wdkv + (long)(n0 + j * 16) * 2048 + k0, 2048);
      acc[j] = wmma_bf16(fa, fb, acc[j]);
    }
  }
  const int nn = lane & 15, mb = (lane >> 4) * 8;
#pragma unroll
  for (int j = 0; j < 4; ++j)
#pragma unroll
    for (int r = 0; r < 8; ++r)
      tile[(mb + r) * 512 + n0 + j * 16 + nn] = acc[j][r];
  __syncthreads();

#pragma unroll
  for (int rr = 0; rr < 2; ++rr) {
    const int row = wave * 2 + rr;
    float s = 0.f, ss = 0.f;
#pragma unroll
    for (int e = 0; e < 16; ++e) {
      const float v = tile[row * 512 + lane * 16 + e];
      s += v;
      ss += v * v;
    }
#pragma unroll
    for (int off = 16; off > 0; off >>= 1) {
      s += __shfl_xor(s, off, 32);
      ss += __shfl_xor(ss, off, 32);
    }
    const float mu = s * (1.f / 512.f);
    const float var = ss * (1.f / 512.f) - mu * mu;
    const float rstd = rsqrtf(var + 1e-5f);
    const long rowg = (long)(m0 + row);
#pragma unroll
    for (int e = 0; e < 16; ++e) {
      const int col = lane * 16 + e;
      const float v = (tile[row * 512 + col] - mu) * rstd * ln_w[col] + ln_b[col];
      c_out[rowg * 512 + col] = v;
      c_bf[rowg * 512 + col] = (__bf16)v;
    }
  }
}

// ---------------------------------------------------------------------------
// Flash attention over the latent cache (one wave = 16-row Q tile).
// ---------------------------------------------------------------------------
__global__ __launch_bounds__(256) void mla_flash_kernel(
    const bf16_t* __restrict__ qlat,  // [B*H, T, 512]
    const bf16_t* __restrict__ cbf,   // [B, S, 512]
    const bf16_t* __restrict__ vT,    // [B, 2048, S]
    bf16_t* __restrict__ ctx,         // [B, T, 2048]
    const int* __restrict__ past_len_p) {
  __shared__ bf16_t pbuf[8][16 * 32];
  const int wave = threadIdx.x >> 5;
  const int lane = threadIdx.x & 31;
  const int gid = blockIdx.x * 8 + wave;
  const int T16 = 2048 / 16;
  const int bh = gid / T16;
  const int t0 = (gid % T16) * 16;
  const int b = bh / 16, h = bh % 16;
  const int pl = *past_len_p;
  const float scale = 0.08838834764831845f;  // 1/sqrt(128)

  v16bf qf[16];
  const bf16_t* qbase = qlat + ((long)bh * 2048 + t0) * 512;
#pragma unroll
  for (int k = 0; k < 16; ++k) qf[k] = load_a_frag(qbase + k * 32, 512);

  v8f out[8] = {};
  float mrow[8], lrow[8];
#pragma unroll
  for (int r = 0; r < 8; ++r) { mrow[r] = -3.0e38f; lrow[r] = 0.f; }

  const int nn = lane & 15, mb = (lane >> 4) * 8;
  int smax = t0 + 15 + pl;
  if (smax > 2047) smax = 2047;
  bf16_t* pw = pbuf[wave];

  for (int s0 = 0; s0 <= smax; s0 += 32) {
    v8f sc0{}, sc1{};
    const bf16_t* kb0 = cbf + ((long)b * 2048 + s0) * 512;
#pragma unroll
    for (int k = 0; k < 16; ++k)
      sc0 = wmma_bf16(qf[k], load_bt_frag(kb0 + k * 32, 512), sc0);
    if (s0 + 16 <= smax) {  // wave-uniform
      const bf16_t* kb1 = kb0 + 16 * 512;
#pragma unroll
      for (int k = 0; k < 16; ++k)
        sc1 = wmma_bf16(qf[k], load_bt_frag(kb1 + k * 32, 512), sc1);
    }
    float alpha[8], p0[8], p1[8];
#pragma unroll
    for (int r = 0; r < 8; ++r) {
      const int t = t0 + mb + r;
      const bool k0m = (s0 + nn) - t > pl;
      const bool k1m = (s0 + 16 + nn) - t > pl;
      const float v0 = k0m ? -3.0e38f : sc0[r] * scale;
      const float v1 = k1m ? -3.0e38f : sc1[r] * scale;
      float vmax = fmaxf(v0, v1);
#pragma unroll
      for (int off = 1; off < 16; off <<= 1)
        vmax = fmaxf(vmax, __shfl_xor(vmax, off, 32));
      const float mnew = fmaxf(mrow[r], vmax);
      const float a = __expf(mrow[r] - mnew);
      const float e0 = k0m ? 0.f : __expf(v0 - mnew);
      const float e1 = k1m ? 0.f : __expf(v1 - mnew);
      float rs = e0 + e1;
#pragma unroll
      for (int off = 1; off < 16; off <<= 1) rs += __shfl_xor(rs, off, 32);
      lrow[r] = lrow[r] * a + rs;
      mrow[r] = mnew;
      alpha[r] = a;
      p0[r] = e0;
      p1[r] = e1;
    }
#pragma unroll
    for (int r = 0; r < 8; ++r) {
      pw[(mb + r) * 32 + nn] = (__bf16)p0[r];
      pw[(mb + r) * 32 + 16 + nn] = (__bf16)p1[r];
    }
    const v16bf pf = load_a_frag((const bf16_t*)pw, 32);
    const bf16_t* vb = vT + ((long)b * 2048 + h * 128) * 2048 + s0;
#pragma unroll
    for (int j = 0; j < 8; ++j) {
#pragma unroll
      for (int r = 0; r < 8; ++r) out[j][r] *= alpha[r];
      out[j] = wmma_bf16(pf, load_bt_frag(vb + (long)(j * 16) * 2048, 2048), out[j]);
    }
  }
#pragma unroll
  for (int j = 0; j < 8; ++j)
#pragma unroll
    for (int r = 0; r < 8; ++r) {
      const long row = (long)b * 2048 + t0 + mb + r;
      ctx[row * 2048 + h * 128 + j * 16 + nn] = (__bf16)(out[j][r] / lrow[r]);
    }
}

// ---------------------------------------------------------------------------
// Host-side launcher
// ---------------------------------------------------------------------------
extern "C" void kernel_launch(void* const* d_in, const int* in_sizes, int n_in,
                              void* d_out, int out_size, void* d_ws, size_t ws_size,
                              hipStream_t stream) {
  (void)in_sizes; (void)n_in; (void)out_size; (void)ws_size;
  const float* x = (const float*)d_in[0];
  const float* W_q = (const float*)d_in[1];
  const float* W_dkv = (const float*)d_in[2];
  const float* W_uk = (const float*)d_in[3];
  const float* W_uv = (const float*)d_in[4];
  const float* W_o = (const float*)d_in[5];
  const float* ln_w = (const float*)d_in[6];
  const float* ln_b = (const float*)d_in[7];
  const int* plp = (const int*)d_in[8];

  const long B = 2, T = 2048, D = 2048, H = 16, HD = 128, L = 512;
  const long BT = B * T;

  char* wsp = (char*)d_ws;
  auto alloc = [&](long bytes) {
    char* p = wsp;
    wsp += (bytes + 255) & ~255L;
    return p;
  };
  bf16_t* x_bf = (bf16_t*)alloc(BT * D * 2);
  bf16_t* wdkv_bf = (bf16_t*)alloc(L * D * 2);
  bf16_t* wuv_bf = (bf16_t*)alloc(D * L * 2);
  bf16_t* wo_bf = (bf16_t*)alloc(D * D * 2);
  bf16_t* wq_bf = (bf16_t*)alloc(D * D * 2);
  bf16_t* wukT_bf = (bf16_t*)alloc(L * D * 2);      // [L, D] = W_uk^T
  bf16_t* c_bf = (bf16_t*)alloc(BT * L * 2);
  bf16_t* absT = (bf16_t*)alloc(H * L * HD * 2);    // [H][L,HD]
  bf16_t* qlat = (bf16_t*)alloc(B * H * T * L * 2); // [B*H][T,L]
  bf16_t* vT = (bf16_t*)alloc(B * D * T * 2);       // [B][D,S]
  bf16_t* ctx = (bf16_t*)alloc(BT * D * 2);

  float* out_f = (float*)d_out;   // [B,T,D]
  float* c_out = out_f + BT * D;  // [B,T,L]

  // 1) precision conversion
  mla_cvt_bf16_kernel<<<1024, 256, 0, stream>>>(x, x_bf, BT * D);
  mla_cvt_bf16_kernel<<<1024, 256, 0, stream>>>(W_dkv, wdkv_bf, L * D);
  mla_cvt_bf16_kernel<<<1024, 256, 0, stream>>>(W_uv, wuv_bf, D * L);
  mla_cvt_bf16_kernel<<<1024, 256, 0, stream>>>(W_o, wo_bf, D * D);
  mla_cvt_bf16_kernel<<<1024, 256, 0, stream>>>(W_q, wq_bf, D * D);
  mla_transpose_cvt_kernel<<<1024, 256, 0, stream>>>(W_uk, wukT_bf, (int)D, (int)L);

  // 2) c = LN(x @ W_dkv^T) -> c_out (fp32 tuple output) + c_bf
  mla_c_ln_kernel<<<(int)(BT / 16), 256, 0, stream>>>(x_bf, wdkv_bf, ln_w, ln_b,
                                                      c_out, c_bf);

  // 3) absorbed^T per head: absT[h][c,d] = sum_k W_q[h*128+d,k] W_uk[k,c]
  {
    const int tiles = (128 / 32) * (512 / 32);
    dim3 grid((tiles + 7) / 8, (int)H);
    mla_gemm_bt_kernel<<<grid, 256, 0, stream>>>(
        wq_bf, D, 0, 128 * D, wukT_bf, D, 0, 0, absT, HD, 0, L * HD,
        128, 512, 2048, (int)H, 2);
  }
  // 4) q_lat[b,h] = x_head @ absorbed[h]
  {
    const int tiles = (2048 / 32) * (512 / 32);
    dim3 grid((tiles + 7) / 8, (int)(B * H));
    mla_gemm_bt_kernel<<<grid, 256, 0, stream>>>(
        x_bf, D, T * D, HD, absT, HD, 0, L * HD, qlat, L, H * T * L, T * L,
        2048, 512, 128, (int)H, 1);
  }
  // 5) v^T[b][n,s] = sum_c c[b,s,c] W_uv[n,c]  (TDM-staged, transposed store)
  {
    dim3 grid((2048 / 128) * (2048 / 64), (int)B);  // 512 x B
    mla_gemm_tdm_kernel<<<grid, 256, 0, stream>>>(
        c_bf, L, T * L, wuv_bf, L, 0, vT, T, D * T,
        2048, 2048, 512, 2);
  }
  // 6) flash attention -> ctx (bf16)
  mla_flash_kernel<<<(int)(B * H * (T / 16) / 8), 256, 0, stream>>>(
      qlat, c_bf, vT, ctx, plp);

  // 7) out = ctx @ W_o^T (TDM-staged, fp32 tuple output)
  {
    dim3 grid((4096 / 128) * (2048 / 64), 1);  // 1024
    mla_gemm_tdm_kernel<<<grid, 256, 0, stream>>>(
        ctx, D, 0, wo_bf, D, 0, out_f, D, 0,
        4096, 2048, 2048, 0);
  }
}
